// BinaryGPTNeoBlock_49838800503280
// MI455X (gfx1250) — compile-verified, matrix operand order checked by reading, not run on hardware
//
#include <hip/hip_runtime.h>
#include <hip/hip_bf16.h>

// ---------------------------------------------------------------------------
// GPT-Neo block (binary weights) for MI455X / gfx1250, wave32 + WMMA bf16.
// Compute-bound (~515 GFLOP vs ~0.4GB HBM traffic) -> all GEMMs routed
// through v_wmma_f32_16x16x32_bf16; attention is single-pass flash style.
// v2: register double-buffered GEMM K-loop + global_prefetch_b8 lookahead.
// ---------------------------------------------------------------------------

#define DEV_INLINE __device__ __forceinline__

static constexpr int Bb = 2;
static constexpr int Ss = 2048;
static constexpr int Dm = 2048;
static constexpr int Hh = 16;
static constexpr int HD = 128;     // Dm / Hh
static constexpr int FF = 8192;    // 4 * Dm
static constexpr int Mrows = Bb * Ss;  // 4096

typedef __attribute__((ext_vector_type(16))) __bf16 bf16x16;
typedef __attribute__((ext_vector_type(8)))  float  f32x8;

union FragU { bf16x16 v; unsigned u[8]; };

DEV_INLINE f32x8 zero8() {
  f32x8 z;
#pragma unroll
  for (int i = 0; i < 8; ++i) z[i] = 0.0f;
  return z;
}

DEV_INLINE f32x8 wmma_bf16(bf16x16 a, bf16x16 b, f32x8 c) {
  // 8 args: (neg_a, A, neg_b, B, c_mod, C, reuse_a, reuse_b)
  return __builtin_amdgcn_wmma_f32_16x16x32_bf16(false, a, false, b, (short)0, c,
                                                 false, false);
}

// Assemble a 16x32 bf16 WMMA fragment from a row-major region (LDS or global).
// ISA layout (16-bit A 16x32, and symmetric B 32x16): lane = row (mod 16);
// lanes 0-15 hold K {0..7,16..23}, lanes 16-31 hold K {8..15,24..31}; each
// VGPR packs 2 consecutive K values -> contiguous 4-byte reads.
DEV_INLINE bf16x16 load_frag(const __bf16* base, int strideElems, int lane) {
  const int r = lane & 15;
  const int h = lane >> 4;
  const unsigned* p = (const unsigned*)(base + (size_t)r * (size_t)strideElems);
  FragU f;
#pragma unroll
  for (int j = 0; j < 4; ++j) f.u[j] = p[h * 4 + j];        // K = h*8 + 2j
#pragma unroll
  for (int j = 0; j < 4; ++j) f.u[4 + j] = p[8 + h * 4 + j]; // K = 16 + h*8 + 2j
  return f.v;
}

DEV_INLINE float gelu_new(float x) {
  const float c = 0.7978845608028654f;
  return 0.5f * x * (1.0f + tanhf(c * (x + 0.044715f * x * x * x)));
}

// ---------------------------------------------------------------------------
// Weight binarization: w_bf16 = tanh(1.0 * w) * 1.0   (KK = AA = 1)
// ---------------------------------------------------------------------------
__global__ void k_binw(const float* __restrict__ w, __bf16* __restrict__ o, int n) {
  int i = blockIdx.x * 256 + threadIdx.x;
  const int stride = gridDim.x * 256;
  for (; i < n; i += stride) o[i] = (__bf16)tanhf(w[i]);
}

// ---------------------------------------------------------------------------
// LayerNorm over D=2048, one block (256 thr) per row, bf16 output.
// ---------------------------------------------------------------------------
__global__ __launch_bounds__(256) void k_ln(const float* __restrict__ x,
                                            const float* __restrict__ g,
                                            const float* __restrict__ b,
                                            __bf16* __restrict__ y) {
  const int row = blockIdx.x;
  const int t = threadIdx.x;
  const float* xr = x + (size_t)row * Dm;
  float s = 0.f, ss = 0.f;
#pragma unroll
  for (int i = t; i < Dm; i += 256) { float v = xr[i]; s += v; ss += v * v; }
  __shared__ float rs[256], rss[256];
  rs[t] = s; rss[t] = ss;
  __syncthreads();
  for (int o = 128; o > 0; o >>= 1) {
    if (t < o) { rs[t] += rs[t + o]; rss[t] += rss[t + o]; }
    __syncthreads();
  }
  const float mu = rs[0] * (1.0f / Dm);
  const float var = rss[0] * (1.0f / Dm) - mu * mu;
  const float rinv = rsqrtf(var + 1e-5f);
  __bf16* yr = y + (size_t)row * Dm;
#pragma unroll
  for (int i = t; i < Dm; i += 256)
    yr[i] = (__bf16)((xr[i] - mu) * rinv * g[i] + b[i]);
}

// ---------------------------------------------------------------------------
// GEMM: C[M,N] = A[M,K] (bf16) x W[N,K]^T (bf16), fp32 accumulate.
// 128x128 block tile, 8 waves, each wave 32x64 (2x4 WMMA tiles).
// Register double-buffered K-loop: next tile's global loads are issued
// before the WMMA block and only waited on at the next LDS store; two-step
// lookahead prefetch (global_prefetch_b8) warms GL2.
// Epilogue: (+bias) (GELU) (+fp32 residual), output bf16 or fp32.
// ---------------------------------------------------------------------------
template <int HAS_BIAS, int HAS_RES, int DO_GELU, int OUT_F32>
__global__ __launch_bounds__(256) void k_gemm(const __bf16* __restrict__ A,
                                              const __bf16* __restrict__ W,
                                              const float* __restrict__ bias,
                                              const float* __restrict__ res,
                                              void* __restrict__ outp,
                                              int Ndim, int Kdim) {
  constexpr int LDP = 34;  // 32 + 2 pad: 68B row stride, conflict-friendly
  __shared__ __bf16 Asm[128 * LDP];
  __shared__ __bf16 Wsm[128 * LDP];
  const int tid = threadIdx.x;
  const int lane = tid & 31, wave = tid >> 5;
  const int n0 = blockIdx.x * 128, m0 = blockIdx.y * 128;
  const int wm = (wave & 3) * 32;   // wave's M offset in tile
  const int wn = (wave >> 2) * 64;  // wave's N offset in tile

  f32x8 acc[2][4];
#pragma unroll
  for (int i = 0; i < 2; ++i)
#pragma unroll
    for (int j = 0; j < 4; ++j) acc[i][j] = zero8();

  const int srow = tid >> 1;            // 0..127
  const int shalf = (tid & 1) * 16;     // 0 or 16 (bf16 elems)

  // Per-thread source rows (this thread copies the same 16-elem slice of the
  // same row every K-step, at column kb+shalf).
  const __bf16* aRow = A + (size_t)(m0 + srow) * Kdim + shalf;
  const __bf16* wRow = W + (size_t)(n0 + srow) * Kdim + shalf;
  unsigned* const al = (unsigned*)(Asm + srow * LDP + shalf);
  unsigned* const wl = (unsigned*)(Wsm + srow * LDP + shalf);

  // ---- preload first K tile into registers ----
  unsigned ra[8], rw[8];
#pragma unroll
  for (int j = 0; j < 8; ++j) {
    ra[j] = ((const unsigned*)aRow)[j];
    rw[j] = ((const unsigned*)wRow)[j];
  }

  for (int kb = 0; kb < Kdim; kb += 32) {
    // ---- commit current tile registers -> LDS ----
#pragma unroll
    for (int j = 0; j < 8; ++j) { al[j] = ra[j]; wl[j] = rw[j]; }
    __syncthreads();

    // ---- issue next tile's global loads (in flight across the WMMAs) ----
    if (kb + 32 < Kdim) {
      const unsigned* ag = (const unsigned*)(aRow + kb + 32);
      const unsigned* wg = (const unsigned*)(wRow + kb + 32);
#pragma unroll
      for (int j = 0; j < 8; ++j) { ra[j] = ag[j]; rw[j] = wg[j]; }
    }
    // ---- two steps ahead: warm GL2 (global_prefetch_b8) ----
    if (kb + 64 < Kdim) {
      __builtin_prefetch(aRow + kb + 64, 0, 1);
      __builtin_prefetch(wRow + kb + 64, 0, 1);
    }

    // ---- compute current tile from LDS ----
    bf16x16 af[2], wf[4];
#pragma unroll
    for (int i = 0; i < 2; ++i) af[i] = load_frag(Asm + (wm + i * 16) * LDP, LDP, lane);
#pragma unroll
    for (int j = 0; j < 4; ++j) wf[j] = load_frag(Wsm + (wn + j * 16) * LDP, LDP, lane);
#pragma unroll
    for (int i = 0; i < 2; ++i)
#pragma unroll
      for (int j = 0; j < 4; ++j) acc[i][j] = wmma_bf16(af[i], wf[j], acc[i][j]);
    __syncthreads();
  }

  // ---- epilogue: C tile layout -> lane = col, VGPR r = row (+8 for hi half)
  const int cc = lane & 15, ch = lane >> 4;
#pragma unroll
  for (int i = 0; i < 2; ++i) {
#pragma unroll
    for (int j = 0; j < 4; ++j) {
      const int col = n0 + wn + j * 16 + cc;
      const float bv = HAS_BIAS ? bias[col] : 0.0f;
#pragma unroll
      for (int r = 0; r < 8; ++r) {
        const int row = m0 + wm + i * 16 + ch * 8 + r;
        const size_t idx = (size_t)row * Ndim + col;
        float v = acc[i][j][r] + bv;
        if (DO_GELU) v = gelu_new(v);
        if (HAS_RES) v += res[idx];
        if (OUT_F32) ((float*)outp)[idx] = v;
        else         ((__bf16*)outp)[idx] = (__bf16)v;
      }
    }
  }
}

// ---------------------------------------------------------------------------
// Flash attention (causal, unscaled scores per GPT-Neo), one block per
// (b, h, 128 q rows); 8 waves, each owns 16 q rows x full HD=128 output.
// ---------------------------------------------------------------------------
__global__ __launch_bounds__(256) void k_attn(const __bf16* __restrict__ Q,
                                              const __bf16* __restrict__ K,
                                              const __bf16* __restrict__ V,
                                              __bf16* __restrict__ O) {
  constexpr int KLD = 132;  // K tile: [32 keys][128 d] + pad
  constexpr int VLD = 36;   // V tile transposed: [128 d][32 keys] + pad
  constexpr int PLD = 34;   // per-wave P staging: [16 q][32 keys] + pad
  __shared__ __bf16 Ksm[32 * KLD];
  __shared__ __bf16 Vts[128 * VLD];
  __shared__ __bf16 Psm[8 * 16 * PLD];

  const int tid = threadIdx.x, lane = tid & 31, wave = tid >> 5;
  const int b = blockIdx.z, h = blockIdx.y, q0 = blockIdx.x * 128;
  const size_t bo = (size_t)b * Ss * Dm + (size_t)h * HD;
  const __bf16* qp = Q + bo;
  const __bf16* kp = K + bo;
  const __bf16* vp = V + bo;
  __bf16* op = O + bo;

  const int qr0 = q0 + wave * 16;

  // Q fragments (16 rows x HD=128 split into 4 K-chunks), loaded once.
  bf16x16 aq[4];
#pragma unroll
  for (int kk = 0; kk < 4; ++kk)
    aq[kk] = load_frag(qp + (size_t)qr0 * Dm + kk * 32, Dm, lane);

  f32x8 oacc[8];
#pragma unroll
  for (int t = 0; t < 8; ++t) oacc[t] = zero8();
  float mrow[8], lrow[8];
#pragma unroll
  for (int r = 0; r < 8; ++r) { mrow[r] = -1e30f; lrow[r] = 0.0f; }

  const int cc = lane & 15, ch = lane >> 4;
  const int kend = q0 + 128;  // causal: keys <= last q row of this block

  for (int kb0 = 0; kb0 < kend; kb0 += 32) {
    // ---- cooperative staging: K row-major, V transposed (d-major) ----
    {
      const int krow = tid >> 3;            // 0..31 key
      const int seg = (tid & 7) * 16;       // d segment
      const unsigned* kg = (const unsigned*)(kp + (size_t)(kb0 + krow) * Dm + seg);
      unsigned* kl = (unsigned*)(Ksm + krow * KLD + seg);
#pragma unroll
      for (int j = 0; j < 8; ++j) kl[j] = kg[j];
      const __bf16* vg = vp + (size_t)(kb0 + krow) * Dm + seg;
#pragma unroll
      for (int c = 0; c < 16; ++c) Vts[(seg + c) * VLD + krow] = vg[c];
      // warm GL2 for the next K/V tiles (global_prefetch_b8)
      if (kb0 + 32 < kend) {
        __builtin_prefetch(kp + (size_t)(kb0 + 32 + krow) * Dm + seg, 0, 1);
        __builtin_prefetch(vp + (size_t)(kb0 + 32 + krow) * Dm + seg, 0, 1);
      }
    }
    __syncthreads();

    // ---- scores: S[16q x 32k] = Q(16x128) x K^T, fp32, unscaled ----
    f32x8 s0 = zero8(), s1 = zero8();
#pragma unroll
    for (int kk = 0; kk < 4; ++kk) {
      bf16x16 b0 = load_frag(Ksm + 0 * KLD + kk * 32, KLD, lane);   // keys 0-15
      bf16x16 b1 = load_frag(Ksm + 16 * KLD + kk * 32, KLD, lane);  // keys 16-31
      s0 = wmma_bf16(aq[kk], b0, s0);
      s1 = wmma_bf16(aq[kk], b1, s1);
    }

    // ---- online softmax (rows live across 16 lanes of one C VGPR) ----
    const int key0 = kb0 + cc, key1 = kb0 + 16 + cc;
    float corr[8];
#pragma unroll
    for (int r = 0; r < 8; ++r) {
      const int qrow = qr0 + ch * 8 + r;
      float v0 = (key0 <= qrow) ? s0[r] : -1e30f;
      float v1 = (key1 <= qrow) ? s1[r] : -1e30f;
      float mx = fmaxf(v0, v1);
#pragma unroll
      for (int off = 8; off > 0; off >>= 1) mx = fmaxf(mx, __shfl_xor(mx, off, 32));
      const float mnew = fmaxf(mrow[r], mx);
      const float c0 = __expf(mrow[r] - mnew);
      const float p0 = __expf(v0 - mnew);
      const float p1 = __expf(v1 - mnew);
      float ps = p0 + p1;
#pragma unroll
      for (int off = 8; off > 0; off >>= 1) ps += __shfl_xor(ps, off, 32);
      lrow[r] = lrow[r] * c0 + ps;
      mrow[r] = mnew;
      corr[r] = c0;
      // C-layout -> row-major P staging in this wave's private LDS slice
      __bf16* pr = Psm + (wave * 16 + ch * 8 + r) * PLD;
      pr[cc] = (__bf16)p0;
      pr[16 + cc] = (__bf16)p1;
    }
#pragma unroll
    for (int t = 0; t < 8; ++t)
#pragma unroll
      for (int r = 0; r < 8; ++r) oacc[t][r] *= corr[r];

    // within-wave LDS store->load ordering for the P round-trip
    asm volatile("s_wait_dscnt 0" ::: "memory");

    // ---- O += P(16x32) x V(32x128) ----
    bf16x16 pf = load_frag(Psm + wave * 16 * PLD, PLD, lane);
#pragma unroll
    for (int t = 0; t < 8; ++t) {
      bf16x16 vf = load_frag(Vts + (t * 16) * VLD, VLD, lane);  // cols d=t*16..+15
      oacc[t] = wmma_bf16(pf, vf, oacc[t]);
    }
    __syncthreads();
  }

  // ---- normalize and write O (bf16) ----
#pragma unroll
  for (int t = 0; t < 8; ++t)
#pragma unroll
    for (int r = 0; r < 8; ++r) {
      const int row = qr0 + ch * 8 + r;
      const int col = t * 16 + cc;
      op[(size_t)row * Dm + col] = (__bf16)(oacc[t][r] / lrow[r]);
    }
}

// ---------------------------------------------------------------------------
// Host launcher
// ---------------------------------------------------------------------------
extern "C" void kernel_launch(void* const* d_in, const int* in_sizes, int n_in,
                              void* d_out, int out_size, void* d_ws, size_t ws_size,
                              hipStream_t stream) {
  (void)in_sizes; (void)n_in; (void)out_size; (void)ws_size;
  const float* hs    = (const float*)d_in[0];
  const float* ln1_g = (const float*)d_in[1];
  const float* ln1_b = (const float*)d_in[2];
  const float* ln2_g = (const float*)d_in[3];
  const float* ln2_b = (const float*)d_in[4];
  const float* wq    = (const float*)d_in[5];
  const float* wk    = (const float*)d_in[6];
  const float* wv    = (const float*)d_in[7];
  const float* wo    = (const float*)d_in[8];
  const float* bo    = (const float*)d_in[9];
  const float* w_fc  = (const float*)d_in[10];
  const float* b_fc  = (const float*)d_in[11];
  const float* w_pr  = (const float*)d_in[12];
  const float* b_pr  = (const float*)d_in[13];
  float* out = (float*)d_out;

  // ---- workspace carve-out (~302 MB total) ----
  size_t off = 0;
  auto alloc = [&](size_t bytes) -> void* {
    void* p = (char*)d_ws + off;
    off += (bytes + 255) & ~(size_t)255;
    return p;
  };
  __bf16* wq_b = (__bf16*)alloc((size_t)Dm * Dm * 2);
  __bf16* wk_b = (__bf16*)alloc((size_t)Dm * Dm * 2);
  __bf16* wv_b = (__bf16*)alloc((size_t)Dm * Dm * 2);
  __bf16* wo_b = (__bf16*)alloc((size_t)Dm * Dm * 2);
  __bf16* wfc_b = (__bf16*)alloc((size_t)FF * Dm * 2);
  __bf16* wpr_b = (__bf16*)alloc((size_t)Dm * FF * 2);
  __bf16* x1 = (__bf16*)alloc((size_t)Mrows * Dm * 2);
  __bf16* qb = (__bf16*)alloc((size_t)Mrows * Dm * 2);
  __bf16* kb = (__bf16*)alloc((size_t)Mrows * Dm * 2);
  __bf16* vb = (__bf16*)alloc((size_t)Mrows * Dm * 2);
  __bf16* ao = (__bf16*)alloc((size_t)Mrows * Dm * 2);
  float*  h1 = (float*)alloc((size_t)Mrows * Dm * 4);
  __bf16* x2 = (__bf16*)alloc((size_t)Mrows * Dm * 2);
  __bf16* m1 = (__bf16*)alloc((size_t)Mrows * FF * 2);

  // ---- binarize weights to bf16 ----
  k_binw<<<4096, 256, 0, stream>>>(wq, wq_b, Dm * Dm);
  k_binw<<<4096, 256, 0, stream>>>(wk, wk_b, Dm * Dm);
  k_binw<<<4096, 256, 0, stream>>>(wv, wv_b, Dm * Dm);
  k_binw<<<4096, 256, 0, stream>>>(wo, wo_b, Dm * Dm);
  k_binw<<<4096, 256, 0, stream>>>(w_fc, wfc_b, FF * Dm);
  k_binw<<<4096, 256, 0, stream>>>(w_pr, wpr_b, Dm * FF);

  // ---- LN1 -> x1 ----
  k_ln<<<Mrows, 256, 0, stream>>>(hs, ln1_g, ln1_b, x1);

  // ---- QKV projections ----
  const dim3 gqkv(Dm / 128, Mrows / 128);
  k_gemm<0, 0, 0, 0><<<gqkv, 256, 0, stream>>>(x1, wq_b, nullptr, nullptr, qb, Dm, Dm);
  k_gemm<0, 0, 0, 0><<<gqkv, 256, 0, stream>>>(x1, wk_b, nullptr, nullptr, kb, Dm, Dm);
  k_gemm<0, 0, 0, 0><<<gqkv, 256, 0, stream>>>(x1, wv_b, nullptr, nullptr, vb, Dm, Dm);

  // ---- flash attention ----
  k_attn<<<dim3(Ss / 128, Hh, Bb), 256, 0, stream>>>(qb, kb, vb, ao);

  // ---- O projection + bias + residual(hidden_states) -> h1 (fp32) ----
  k_gemm<1, 1, 0, 1><<<gqkv, 256, 0, stream>>>(ao, wo_b, bo, hs, h1, Dm, Dm);

  // ---- LN2 -> x2 ----
  k_ln<<<Mrows, 256, 0, stream>>>(h1, ln2_g, ln2_b, x2);

  // ---- MLP: fc + bias + gelu -> m1 (bf16) ----
  k_gemm<1, 0, 1, 0><<<dim3(FF / 128, Mrows / 128), 256, 0, stream>>>(
      x2, wfc_b, b_fc, nullptr, m1, FF, Dm);

  // ---- MLP: proj + bias + residual(h1) -> out (fp32) ----
  k_gemm<1, 1, 0, 1><<<dim3(Dm / 128, Mrows / 128), 256, 0, stream>>>(
      m1, wpr_b, b_pr, h1, out, Dm, FF);
}
